// PairContrastiveLoss_69174743269985
// MI455X (gfx1250) — matrix operand order repaired and checked by reading, not compiled, not run on hardware
//
#include <hip/hip_runtime.h>

typedef __attribute__((ext_vector_type(16))) __bf16          v16bf;
typedef __attribute__((ext_vector_type(8)))  float           v8f;
typedef __attribute__((ext_vector_type(16))) unsigned short  u16x16;
typedef __attribute__((ext_vector_type(8)))  unsigned short  u16x8;

#define D_HID     768
#define N_PATCH   576
#define N_BATCH   64
#define N_LAT     1024
#define ROWS_TOT  (N_BATCH * N_PATCH)   /* 36864 */
#define TAU_INV   2.0f
#define CS_EPS    1e-8f

// ---------- helpers -------------------------------------------------------

__device__ __forceinline__ unsigned short f2bf(float f) {
  unsigned int u = __float_as_uint(f);
  unsigned int r = u + 0x7FFFu + ((u >> 16) & 1u);   // round-to-nearest-even
  return (unsigned short)(r >> 16);
}

// A fragment (16x32 bf16, MxK): lane&15 = row, lane>>4 selects K-half.
// VGPR0-3 hold K = half*8 .. +7 ; VGPR4-7 hold K = 16+half*8 .. +7
__device__ __forceinline__ v16bf load_frag_a(const unsigned short* base, int ld,
                                             int mbase, int kbase, int lane) {
  int row  = mbase + (lane & 15);
  int half = lane >> 4;
  const unsigned short* p = base + (size_t)row * ld + kbase + half * 8;
  u16x8 lo = *(const u16x8*)(p);
  u16x8 hi = *(const u16x8*)(p + 16);
  u16x16 t = __builtin_shufflevector(lo, hi, 0,1,2,3,4,5,6,7,8,9,10,11,12,13,14,15);
  return __builtin_bit_cast(v16bf, t);
}

// B fragment (32x16 bf16, KxN): lane&15 = column N, lane>>4 selects K-half.
// Source is row-major [N][K] (we multiply by the transpose); same K split as A.
__device__ __forceinline__ v16bf load_frag_b(const unsigned short* base, int ld,
                                             int nbase, int kbase, int lane) {
  int col  = nbase + (lane & 15);
  int half = lane >> 4;
  const unsigned short* p = base + (size_t)col * ld + kbase + half * 8;
  u16x8 lo = *(const u16x8*)(p);
  u16x8 hi = *(const u16x8*)(p + 16);
  u16x16 t = __builtin_shufflevector(lo, hi, 0,1,2,3,4,5,6,7,8,9,10,11,12,13,14,15);
  return __builtin_bit_cast(v16bf, t);
}

__device__ __forceinline__ v8f wmma_bf16(v16bf a, v16bf b, v8f c) {
  return __builtin_amdgcn_wmma_f32_16x16x32_bf16(false, a, false, b,
                                                 (short)0, c, false, false);
}

__device__ __forceinline__ float wave_sum(float v) {
#pragma unroll
  for (int m = 16; m >= 1; m >>= 1) v += __shfl_xor(v, m, 32);
  return v;
}

// ---------- prep kernels --------------------------------------------------

__global__ void k_init(float* sums) {
  if (threadIdx.x < 128) sums[threadIdx.x] = 0.f;
}

__global__ void k_convw(const float* __restrict__ w, unsigned short* __restrict__ wb) {
  int i = blockIdx.x * blockDim.x + threadIdx.x;   // 768*768 exact
  wb[i] = f2bf(w[i]);
}

__global__ void k_latnorm(const float* __restrict__ lat, unsigned short* __restrict__ lb) {
  int lane = threadIdx.x & 31, wave = threadIdx.x >> 5;
  int l = blockIdx.x * 8 + wave;                   // 1024 rows
  const float* p = lat + (size_t)l * D_HID;
  float s = 0.f;
  for (int i = lane; i < D_HID; i += 32) { float x = p[i]; s += x * x; }
  s = wave_sum(s);
  float inv = 1.f / fmaxf(sqrtf(s), CS_EPS);
  for (int i = lane; i < D_HID; i += 32) lb[(size_t)l * D_HID + i] = f2bf(p[i] * inv);
}

// im2col gather: A[row = b*576 + ph*24+pw][k = c*256 + r*16 + col] = x[b,c,ph*16+r,pw*16+col]
__global__ void k_patch(const float* __restrict__ x, unsigned short* __restrict__ A) {
  long long g = (long long)blockIdx.x * blockDim.x + threadIdx.x;  // 7,077,888 groups of 4
  int k4 = (int)(g % 192) * 4;
  long long row = g / 192;
  int b = (int)(row / N_PATCH);
  int n = (int)(row % N_PATCH);
  int ph = n / 24, pw = n % 24;
  int c = k4 >> 8, rem = k4 & 255, r = rem >> 4, col = rem & 15;
  size_t xi = ((size_t)(b * 3 + c) * 384 + (size_t)(ph * 16 + r)) * 384 + pw * 16 + col;
  float4 v = *(const float4*)(x + xi);
  unsigned long long pk =
      (unsigned long long)f2bf(v.x)
    | ((unsigned long long)f2bf(v.y) << 16)
    | ((unsigned long long)f2bf(v.z) << 32)
    | ((unsigned long long)f2bf(v.w) << 48);
  __builtin_nontemporal_store(pk, (unsigned long long*)(A + (size_t)row * D_HID + k4));
}

// ---------- patch-embed GEMM + bias + row L2-normalize --------------------
// Block = 16 output rows x 768 cols. 16 waves, 3 N-tiles per wave, K=768 in 24 WMMA steps.
__global__ void __launch_bounds__(512)
k_embed(const unsigned short* __restrict__ A, const unsigned short* __restrict__ Wb,
        const float* __restrict__ bias, unsigned short* __restrict__ pn) {
  __shared__ float rn[16];
  int lane = threadIdx.x & 31, wave = threadIdx.x >> 5;
  int half = lane >> 4;
  int mbase = blockIdx.x * 16;
  if (threadIdx.x < 16) rn[threadIdx.x] = 0.f;
  __syncthreads();

  int n0 = (wave * 3 + 0) * 16, n1 = (wave * 3 + 1) * 16, n2 = (wave * 3 + 2) * 16;
  v8f acc0 = {}, acc1 = {}, acc2 = {};
  for (int kk = 0; kk < D_HID; kk += 32) {
    if (kk + 32 < D_HID)
      __builtin_prefetch(A + (size_t)(mbase + (lane & 15)) * D_HID + kk + 32, 0, 1);
    v16bf a = load_frag_a(A, D_HID, mbase, kk, lane);
    acc0 = wmma_bf16(a, load_frag_b(Wb, D_HID, n0, kk, lane), acc0);
    acc1 = wmma_bf16(a, load_frag_b(Wb, D_HID, n1, kk, lane), acc1);
    acc2 = wmma_bf16(a, load_frag_b(Wb, D_HID, n2, kk, lane), acc2);
  }
  float b0 = bias[n0 + (lane & 15)], b1 = bias[n1 + (lane & 15)], b2 = bias[n2 + (lane & 15)];
  float s[8];
#pragma unroll
  for (int v = 0; v < 8; ++v) {
    acc0[v] += b0; acc1[v] += b1; acc2[v] += b2;
    s[v] = acc0[v] * acc0[v] + acc1[v] * acc1[v] + acc2[v] * acc2[v];
  }
  // reduce across the 16 lanes sharing a row (xor masks keep lane bit4 fixed)
#pragma unroll
  for (int m = 1; m <= 8; m <<= 1) {
#pragma unroll
    for (int v = 0; v < 8; ++v) s[v] += __shfl_xor(s[v], m, 32);
  }
  if ((lane & 15) == 0) {
#pragma unroll
    for (int v = 0; v < 8; ++v) atomicAdd(&rn[v + 8 * half], s[v]);
  }
  __syncthreads();
#pragma unroll
  for (int v = 0; v < 8; ++v) {
    int rloc = v + 8 * half;
    float inv = 1.f / fmaxf(sqrtf(rn[rloc]), CS_EPS);
    size_t rowoff = (size_t)(mbase + rloc) * D_HID;
    pn[rowoff + n0 + (lane & 15)] = f2bf(acc0[v] * inv);
    pn[rowoff + n1 + (lane & 15)] = f2bf(acc1[v] * inv);
    pn[rowoff + n2 + (lane & 15)] = f2bf(acc2[v] * inv);
  }
}

// ---------- all-pairs cosine sim: sum exp(SIM/tau) off-diagonal -----------
// 2x2 register blocking: each wave computes a 32x32 output block.
__global__ void __launch_bounds__(256)
k_sim(const unsigned short* __restrict__ pn, float* __restrict__ far) {
  int lane = threadIdx.x & 31, wave = threadIdx.x >> 5;
  int tile = blockIdx.x * 8 + wave;        // 64 batches * 18*18 blocks
  int b  = tile / 324;
  int t  = tile % 324;
  int ti = t / 18, tj = t % 18;
  const unsigned short* pb = pn + (size_t)b * N_PATCH * D_HID;
  int r0 = ti * 32, r1 = r0 + 16, c0 = tj * 32, c1 = c0 + 16;
  v8f a00 = {}, a01 = {}, a10 = {}, a11 = {};
  for (int kk = 0; kk < D_HID; kk += 32) {
    v16bf fa0 = load_frag_a(pb, D_HID, r0, kk, lane);
    v16bf fa1 = load_frag_a(pb, D_HID, r1, kk, lane);
    v16bf fb0 = load_frag_b(pb, D_HID, c0, kk, lane);
    v16bf fb1 = load_frag_b(pb, D_HID, c1, kk, lane);
    a00 = wmma_bf16(fa0, fb0, a00);
    a01 = wmma_bf16(fa0, fb1, a01);
    a10 = wmma_bf16(fa1, fb0, a10);
    a11 = wmma_bf16(fa1, fb1, a11);
  }
  int half = lane >> 4;
  int col0 = c0 + (lane & 15), col1 = c1 + (lane & 15);
  float s = 0.f;
#pragma unroll
  for (int v = 0; v < 8; ++v) {
    int row0 = r0 + v + 8 * half, row1 = r1 + v + 8 * half;
    s += (row0 == col0) ? 0.f : __expf(a00[v] * TAU_INV);
    s += (row0 == col1) ? 0.f : __expf(a01[v] * TAU_INV);
    s += (row1 == col0) ? 0.f : __expf(a10[v] * TAU_INV);
    s += (row1 == col1) ? 0.f : __expf(a11[v] * TAU_INV);
  }
  s = wave_sum(s);
  if (lane == 0) atomicAdd(&far[b], s);
}

// ---------- patch vs latent: sum exp(close/tau) ---------------------------
// 2x4 register blocking: each wave computes a 32x64 output block.
__global__ void __launch_bounds__(256)
k_close(const unsigned short* __restrict__ pn, const unsigned short* __restrict__ lb,
        float* __restrict__ close) {
  int lane = threadIdx.x & 31, wave = threadIdx.x >> 5;
  int tile = blockIdx.x * 8 + wave;        // 1152 row-groups * 16 col-groups
  int rg = tile >> 4;
  int cg = tile & 15;
  int r0 = rg * 32, r1 = r0 + 16, cb = cg * 64;
  v8f acc[8] = {};                         // [m*4 + n], m in 0..1, n in 0..3
  for (int kk = 0; kk < D_HID; kk += 32) {
    v16bf fa0 = load_frag_a(pn, D_HID, r0, kk, lane);
    v16bf fa1 = load_frag_a(pn, D_HID, r1, kk, lane);
#pragma unroll
    for (int j = 0; j < 4; ++j) {
      v16bf fb = load_frag_b(lb, D_HID, cb + j * 16, kk, lane);
      acc[j]     = wmma_bf16(fa0, fb, acc[j]);
      acc[4 + j] = wmma_bf16(fa1, fb, acc[4 + j]);
    }
  }
  float s = 0.f;
#pragma unroll
  for (int t8 = 0; t8 < 8; ++t8) {
#pragma unroll
    for (int v = 0; v < 8; ++v) s += __expf(acc[t8][v] * TAU_INV);
  }
  s = wave_sum(s);
  int bidx = r0 / N_PATCH;                 // 32 rows share one batch (576 % 32 == 0)
  if (lane == 0) atomicAdd(&close[bidx], s);
}

// ---------- finalize ------------------------------------------------------
__global__ void k_final(const float* __restrict__ far, const float* __restrict__ close,
                        float* __restrict__ out) {
  __shared__ float sh[64];
  int t = threadIdx.x;
  sh[t] = -(logf(close[t]) - logf(far[t]));
  __syncthreads();
  if (t == 0) {
    float s = 0.f;
    for (int i = 0; i < 64; ++i) s += sh[i];
    out[0] = s / 64.f;
  }
}

// ---------- launcher ------------------------------------------------------

extern "C" void kernel_launch(void* const* d_in, const int* in_sizes, int n_in,
                              void* d_out, int out_size, void* d_ws, size_t ws_size,
                              hipStream_t stream) {
  const float* x      = (const float*)d_in[0];   // (64,3,384,384)
  const float* conv_w = (const float*)d_in[1];   // (768,3,16,16) == (768,768)
  const float* conv_b = (const float*)d_in[2];   // (768,)
  const float* latent = (const float*)d_in[3];   // (1,1024,768)
  float* out = (float*)d_out;

  char* ws = (char*)d_ws;
  const size_t szA  = (size_t)ROWS_TOT * D_HID * 2;   // 56,623,104
  const size_t szW  = (size_t)D_HID * D_HID * 2;      //  1,179,648
  const size_t szL  = (size_t)N_LAT * D_HID * 2;      //  1,572,864
  const size_t szP  = szA;
  unsigned short* A  = (unsigned short*)(ws);
  unsigned short* Wb = (unsigned short*)(ws + szA);
  unsigned short* Lb = (unsigned short*)(ws + szA + szW);
  unsigned short* Pn = (unsigned short*)(ws + szA + szW + szL);
  float* sums  = (float*)(ws + szA + szW + szL + szP);
  float* far   = sums;
  float* close = sums + 64;

  k_init   <<<1,     128, 0, stream>>>(sums);
  k_convw  <<<2304,  256, 0, stream>>>(conv_w, Wb);
  k_latnorm<<<128,   256, 0, stream>>>(latent, Lb);
  k_patch  <<<27648, 256, 0, stream>>>(x, A);
  k_embed  <<<2304,  512, 0, stream>>>(A, Wb, conv_b, Pn);
  k_sim    <<<2592,  256, 0, stream>>>(Pn, far);
  k_close  <<<2304,  256, 0, stream>>>(Pn, Lb, close);
  k_final  <<<1,      64, 0, stream>>>(far, close, out);
}
